// RobustCTRNN_46866683134473
// MI455X (gfx1250) — compile-verified, matrix operand order checked by reading, not run on hardware
//
#include <hip/hip_runtime.h>
#include <hip/hip_bf16.h>

typedef __attribute__((ext_vector_type(16))) _Float16 v16h;
typedef __attribute__((ext_vector_type(8)))  _Float16 v8h;
typedef __attribute__((ext_vector_type(8)))  float    v8f;
typedef __attribute__((ext_vector_type(4)))  float    v4f;

#define T_STEPS 1000
#define B_SZ    512
#define I_DIM   64
#define H_DIM   64
#define F_DIM   128
#define KEEP    5
#define ALPHA_C 0.2f

// ---- fragment helpers (ISA 7.12.2 16-bit layouts) ----
// 16-bit A (16x32) / B-as-[N][K]: lane L -> row/col (L&15); two 8-half runs:
//   k0 = kbase + (L>>4)*8   and   k0 + 16
__device__ __forceinline__ v16h frag_from_lds(const _Float16* base, int ld,
                                              int lane, int kbase) {
  int r  = lane & 15;
  int k0 = kbase + ((lane >> 4) << 3);
  const _Float16* p = base + r * ld + k0;
  v8h lo = *(const v8h*)(p);
  v8h hi = *(const v8h*)(p + 16);
  return __builtin_shufflevector(lo, hi, 0,1,2,3,4,5,6,7,8,9,10,11,12,13,14,15);
}

// raw f32 A-frag gather for x_t (two frags, K=64): 8 x float4 per lane
__device__ __forceinline__ void load_x_raw(v4f xr[8], const float* xtile, int lane) {
  int r    = lane & 15;
  int koff = ((lane >> 4) << 3);
  const float* p = xtile + r * I_DIM + koff;
  xr[0] = *(const v4f*)(p);
  xr[1] = *(const v4f*)(p + 4);
  xr[2] = *(const v4f*)(p + 16);
  xr[3] = *(const v4f*)(p + 20);
  xr[4] = *(const v4f*)(p + 32);
  xr[5] = *(const v4f*)(p + 36);
  xr[6] = *(const v4f*)(p + 48);
  xr[7] = *(const v4f*)(p + 52);
}

__device__ __forceinline__ v16h cvt_frag(const v4f* q) {
  v16h a;
#pragma unroll
  for (int i = 0; i < 4; ++i) {
    a[i]      = (_Float16)q[0][i];
    a[4 + i]  = (_Float16)q[1][i];
    a[8 + i]  = (_Float16)q[2][i];
    a[12 + i] = (_Float16)q[3][i];
  }
  return a;
}

__device__ __forceinline__ v8f wmma_f16(v16h a, v16h b, v8f c) {
  return __builtin_amdgcn_wmma_f32_16x16x32_f16(false, a, false, b,
                                                (short)0, c, false, false);
}

#define SMEM_BYTES (64*64*2 /*Win*/ + 64*64*2 /*Whh*/ + 128*320*2 /*W1*/ + \
                    64*128*2 /*W2*/ + KEEP*16*64*2 /*ring*/ + 16*128*2 /*mlp*/ + \
                    (64+128+64)*4 /*biases*/)

__global__ void __launch_bounds__(256, 1)
ctrnn_persistent_kernel(const float* __restrict__ x,   const float* __restrict__ h0,
                        const float* __restrict__ W_in, const float* __restrict__ b_in,
                        const float* __restrict__ W_hh, const float* __restrict__ b_hh,
                        const float* __restrict__ W1,   const float* __restrict__ b1,
                        const float* __restrict__ W2,   const float* __restrict__ b2,
                        float* __restrict__ out) {
  extern __shared__ char smem[];
  _Float16* sWin = (_Float16*)smem;           // [64][64]
  _Float16* sWhh = sWin + 64 * 64;            // [64][64]
  _Float16* sW1  = sWhh + 64 * 64;            // [128][320]
  _Float16* sW2  = sW1 + 128 * 320;           // [64][128]
  _Float16* ring = sW2 + 64 * 128;            // [KEEP][16][64]
  _Float16* mlpb = ring + KEEP * 16 * 64;     // [16][128]
  float* sbinh = (float*)(mlpb + 16 * 128);   // [64]  b_in + b_hh
  float* sb1   = sbinh + 64;                  // [128]
  float* sb2   = sb1 + 128;                   // [64]

  const int tid   = threadIdx.x;
  const int lane  = tid & 31;
  const int w     = tid >> 5;          // 8 waves
  const int bbase = blockIdx.x * 16;   // 16 batch rows per workgroup
  const bool isH  = (w < 4);           // waves 0-3: hidden path; 4-7: MLP path

  // ---- one-time: stage weights as f16 in LDS ----
  for (int i = tid; i < 64 * 64; i += 256) {
    sWin[i] = (_Float16)W_in[i];
    sWhh[i] = (_Float16)W_hh[i];
  }
  for (int i = tid; i < 128 * 320; i += 256) sW1[i] = (_Float16)W1[i];
  for (int i = tid; i < 64 * 128;  i += 256) sW2[i] = (_Float16)W2[i];
  for (int i = tid; i < 64;  i += 256) { sbinh[i] = b_in[i] + b_hh[i]; sb2[i] = b2[i]; }
  for (int i = tid; i < 128; i += 256) sb1[i] = b1[i];
  for (int i = tid; i < 4 * 16 * 64; i += 256) ring[i] = (_Float16)0.f;
  // seed ring slot 4 with h0 (= h_{-1}); slots 0-3 zero (buf0 zeros)
  for (int i = tid; i < 16 * 64; i += 256) {
    int m = i >> 6, c = i & 63;
    ring[(4 * 16 + m) * 64 + c] = (_Float16)h0[(size_t)(bbase + m) * H_DIM + c];
  }
  __syncthreads();

  // ---- per-role register-resident state ----
  v16h winb[2], whhb[2], w2b[4];   // H-waves
  v16h w1b0[10], w1b1[10];         // F-waves: W1 B-frags for its 2 F-tiles (160 VGPRs)
  v8f  hfrag = {};
  float binh = 0.f, b2v = 0.f, b1a = 0.f, b1b = 0.f;

  if (isH) {
    const int hw = w;
#pragma unroll
    for (int j = 0; j < 2; ++j) {
      winb[j] = frag_from_lds(sWin + (hw * 16) * 64, 64, lane, j * 32);
      whhb[j] = frag_from_lds(sWhh + (hw * 16) * 64, 64, lane, j * 32);
    }
#pragma unroll
    for (int j = 0; j < 4; ++j)
      w2b[j] = frag_from_lds(sW2 + (hw * 16) * 128, 128, lane, j * 32);
    binh = sbinh[hw * 16 + (lane & 15)];
    b2v  = sb2[hw * 16 + (lane & 15)];
    {
      int col = hw * 16 + (lane & 15);
#pragma unroll
      for (int r = 0; r < 8; ++r) {
        int m = r + ((lane >> 4) << 3);
        hfrag[r] = h0[(size_t)(bbase + m) * H_DIM + col];
      }
    }
  } else {
    const int fw = w - 4;
#pragma unroll
    for (int j = 0; j < 10; ++j) {
      w1b0[j] = frag_from_lds(sW1 + (2 * fw)     * 16 * 320, 320, lane, j * 32);
      w1b1[j] = frag_from_lds(sW1 + (2 * fw + 1) * 16 * 320, 320, lane, j * 32);
    }
    b1a = sb1[(2 * fw) * 16 + (lane & 15)];
    b1b = sb1[(2 * fw + 1) * 16 + (lane & 15)];
  }

  int head = 0;  // oldest ring slot
  const float* xt   = x   + (size_t)bbase * I_DIM;
  float*       outt = out + (size_t)bbase * H_DIM;

  v4f xr[8];
  if (isH) load_x_raw(xr, xt, lane);   // prefetch x_0

  for (int t = 0; t < T_STEPS; ++t) {
    v8f hnew;
    if (isH) {
      // ---- h_obs = relu(x W_in^T + h W_hh^T + b); runs parallel to MLP waves
      int newest = head + 4; if (newest >= KEEP) newest -= KEEP;
      const _Float16* hslot = ring + newest * 16 * 64;
      v16h ha0 = frag_from_lds(hslot, 64, lane, 0);
      v16h ha1 = frag_from_lds(hslot, 64, lane, 32);
      v16h xa0 = cvt_frag(&xr[0]);
      v16h xa1 = cvt_frag(&xr[4]);
      if (t + 1 < T_STEPS)                  // prefetch next step's x
        load_x_raw(xr, xt + (size_t)B_SZ * I_DIM, lane);
      v8f acc;
#pragma unroll
      for (int r = 0; r < 8; ++r) acc[r] = binh;
      acc = wmma_f16(xa0, winb[0], acc);
      acc = wmma_f16(xa1, winb[1], acc);
      acc = wmma_f16(ha0, whhb[0], acc);
      acc = wmma_f16(ha1, whhb[1], acc);
#pragma unroll
      for (int r = 0; r < 8; ++r) {
        float ho = acc[r] > 0.f ? acc[r] : 0.f;
        hnew[r] = hfrag[r] * (1.f - ALPHA_C) + ho * ALPHA_C;
      }
    } else if (t >= KEEP) {
      // ---- MLP hidden: [16,320] @ W1^T -> relu -> f16 stage (software-pipelined)
      v8f m0, m1;
#pragma unroll
      for (int r = 0; r < 8; ++r) { m0[r] = b1a; m1[r] = b1b; }
      v16h ma = frag_from_lds(ring + head * 16 * 64, 64, lane, 0);
#pragma unroll
      for (int j = 0; j < 10; ++j) {
        v16h manext = ma;
        if (j < 9) {
          int jn = j + 1;
          int slot = head + (jn >> 1); if (slot >= KEEP) slot -= KEEP;
          manext = frag_from_lds(ring + slot * 16 * 64, 64, lane, (jn & 1) * 32);
        }
        m0 = wmma_f16(ma, w1b0[j], m0);
        m1 = wmma_f16(ma, w1b1[j], m1);
        ma = manext;
      }
      const int fw = w - 4;
      int col0 = (2 * fw) * 16 + (lane & 15);
#pragma unroll
      for (int r = 0; r < 8; ++r) {
        int m = r + ((lane >> 4) << 3);
        float a0 = m0[r] > 0.f ? m0[r] : 0.f;
        float a1 = m1[r] > 0.f ? m1[r] : 0.f;
        mlpb[m * F_DIM + col0]      = (_Float16)a0;
        mlpb[m * F_DIM + col0 + 16] = (_Float16)a1;
      }
    }

    // mlpb published; also: every ring read of this step is now complete
    __syncthreads();

    if (isH) {
      if (t >= KEEP) {
        // ---- h_mlp tile: [16,128] @ W2^T (pipelined), then 0.5/0.5 mix
        v8f oacc;
#pragma unroll
        for (int r = 0; r < 8; ++r) oacc[r] = b2v;
        v16h aa = frag_from_lds(mlpb, F_DIM, lane, 0);
#pragma unroll
        for (int j = 0; j < 4; ++j) {
          v16h anext = aa;
          if (j < 3) anext = frag_from_lds(mlpb, F_DIM, lane, (j + 1) * 32);
          oacc = wmma_f16(aa, w2b[j], oacc);
          aa = anext;
        }
#pragma unroll
        for (int r = 0; r < 8; ++r) hnew[r] = 0.5f * hnew[r] + 0.5f * oacc[r];
      }
      // push h_t over the (already consumed) oldest slot + stream out[t]
      int col = w * 16 + (lane & 15);
      _Float16* dst = ring + head * 16 * 64;
#pragma unroll
      for (int r = 0; r < 8; ++r) {
        int m = r + ((lane >> 4) << 3);
        dst[m * 64 + col] = (_Float16)hnew[r];
        outt[(size_t)m * H_DIM + col] = hnew[r];
      }
      hfrag = hnew;
    }

    head += 1; if (head >= KEEP) head = 0;
    xt   += (size_t)B_SZ * I_DIM;
    outt += (size_t)B_SZ * H_DIM;
    __syncthreads();  // ring/mlpb consumable next step
  }

  // ---- hT (second output, concatenated after out) ----
  if (isH) {
    float* hT = out + (size_t)T_STEPS * B_SZ * H_DIM;
    int col = w * 16 + (lane & 15);
#pragma unroll
    for (int r = 0; r < 8; ++r) {
      int m = r + ((lane >> 4) << 3);
      hT[(size_t)(bbase + m) * H_DIM + col] = hfrag[r];
    }
  }
}

extern "C" void kernel_launch(void* const* d_in, const int* in_sizes, int n_in,
                              void* d_out, int out_size, void* d_ws, size_t ws_size,
                              hipStream_t stream) {
  (void)in_sizes; (void)n_in; (void)d_ws; (void)ws_size; (void)out_size;
  const float* x    = (const float*)d_in[0];
  const float* h0   = (const float*)d_in[1];
  const float* W_in = (const float*)d_in[2];
  const float* b_in = (const float*)d_in[3];
  const float* W_hh = (const float*)d_in[4];
  const float* b_hh = (const float*)d_in[5];
  const float* W1   = (const float*)d_in[6];
  const float* b1   = (const float*)d_in[7];
  const float* W2   = (const float*)d_in[8];
  const float* b2   = (const float*)d_in[9];
  float* out = (float*)d_out;

  // allow >64KB dynamic LDS (weights + ring fully WGP-resident: ~127 KB)
  (void)hipFuncSetAttribute((const void*)ctrnn_persistent_kernel,
                            hipFuncAttributeMaxDynamicSharedMemorySize,
                            SMEM_BYTES);

  dim3 grid(B_SZ / 16);   // 32 workgroups, one 16-row batch tile each
  dim3 block(256);        // 8 waves: 4 hidden-path + 4 MLP-path (W1 in VGPRs)
  ctrnn_persistent_kernel<<<grid, block, SMEM_BYTES, stream>>>(
      x, h0, W_in, b_in, W_hh, b_hh, W1, b1, W2, b2, out);
}
